// GlobalDensityApprox_45732811768425
// MI455X (gfx1250) — matrix operand order repaired and checked by reading, not compiled, not run on hardware
//
#include <hip/hip_runtime.h>
#include <hip/hip_bf16.h>
#include <stdint.h>

// ---------------------------------------------------------------------------
// Types for CDNA5 WMMA (wave32, 16x16x32 bf16 -> f32)
// ---------------------------------------------------------------------------
typedef __attribute__((ext_vector_type(8)))  float  v8f;
typedef __attribute__((ext_vector_type(16))) __bf16 v16bf;

union BF16Frag { v16bf v; uint4 q[2]; };
union F32x8    { v8f v; float f[8]; };

__device__ inline uint16_t f2bf(float x) {
  uint32_t u = __float_as_uint(x);
  uint32_t r = u + 0x7FFFu + ((u >> 16) & 1u);
  return (uint16_t)(r >> 16);
}
__device__ inline float bf2f(uint16_t h) { return __uint_as_float(((uint32_t)h) << 16); }

__device__ inline float wave_sum(float v) {
  #pragma unroll
  for (int m = 16; m > 0; m >>= 1) v += __shfl_xor(v, m, 32);
  return v;
}
__device__ inline float wave_max(float v) {
  #pragma unroll
  for (int m = 16; m > 0; m >>= 1) v = fmaxf(v, __shfl_xor(v, m, 32));
  return v;
}
__device__ inline float softplusf(float x) { return (x > 20.f) ? x : log1pf(__expf(x)); }

// ---------------------------------------------------------------------------
// Generic GEMM: C[M,N] = act(A[M,K] @ W[N,K]^T + bias) (* mul)  -> f32/bf16
//
// Block = 128 threads = 4 waves. Block tile = 128 rows x 32 cols.
// Each wave computes a 32x32 output (2x2 WMMA tiles, 4 accumulators) so each
// A/B fragment feeds two v_wmma_f32_16x16x32_bf16 ops.
// The 32x32 B (weight) tile is shared by all 4 waves: staged into LDS with
// gfx1250 async global->LDS copies (ASYNCcnt), double buffered, and read back
// as fragments with ds_load_b128. A stream uses direct b128 loads + prefetch.
// ---------------------------------------------------------------------------
__global__ __launch_bounds__(128) void gemm_bf16_wmma(
    const uint16_t* __restrict__ A, const uint16_t* __restrict__ W,
    const float* __restrict__ bias, const uint16_t* __restrict__ mul,
    float* __restrict__ outF, uint16_t* __restrict__ outB,
    int M, int N, int K, int act)
{
  __shared__ uint16_t Bs[2][32 * 32];   // [buf][col(32) x k(32)]

  const int tid  = threadIdx.x;
  const int lane = tid & 31;
  const int wv   = tid >> 5;
  const int l16  = lane & 15;
  const int hi   = lane >> 4;
  const int mbase = (blockIdx.y * 4 + wv) * 32;  // this wave's 32 rows
  const int nbase = blockIdx.x * 32;             // this block's 32 cols

  // A fragment row pointers for the wave's two 16-row tiles
  const uint16_t* arow0 = A + (size_t)(mbase + l16) * K + hi * 8;
  const uint16_t* arow1 = arow0 + (size_t)16 * K;

  // async B staging: each thread copies 16B; col = tid>>2, k-chunk = (tid&3)*8
  const int bn = tid >> 2;
  const int bk = (tid & 3) * 8;
  const uint16_t* bsrc = W + (size_t)(nbase + bn) * K + bk;

  F32x8 acc[2][2];
  #pragma unroll
  for (int i = 0; i < 2; i++)
    #pragma unroll
    for (int j = 0; j < 2; j++)
      #pragma unroll
      for (int r = 0; r < 8; r++) acc[i][j].f[r] = 0.f;

  auto issueB = [&](int buf, int k0) {
    uint32_t lds = (uint32_t)(uintptr_t)(&Bs[buf][bn * 32 + bk]);
    uint64_t ga  = (uint64_t)(uintptr_t)(bsrc + k0);
    asm volatile("global_load_async_to_lds_b128 %0, %1, off"
                 :: "v"(lds), "v"(ga) : "memory");
  };

  issueB(0, 0);
  int cur = 0;
  for (int k0 = 0; k0 < K; k0 += 32) {
    const bool more = (k0 + 32) < K;
    if (more) {
      issueB(cur ^ 1, k0 + 32);
      __builtin_prefetch(arow0 + k0 + 32, 0, 3);
      __builtin_prefetch(arow1 + k0 + 32, 0, 3);
      asm volatile("s_wait_asynccnt 0x1" ::: "memory");
    } else {
      asm volatile("s_wait_asynccnt 0x0" ::: "memory");
    }
    __syncthreads();   // B tile visible to all 4 waves

    BF16Frag a0, a1, b0, b1;
    a0.q[0] = *(const uint4*)(arow0 + k0);
    a0.q[1] = *(const uint4*)(arow0 + k0 + 16);
    a1.q[0] = *(const uint4*)(arow1 + k0);
    a1.q[1] = *(const uint4*)(arow1 + k0 + 16);
    const uint16_t* bs0 = &Bs[cur][(size_t)l16 * 32 + hi * 8];         // cols 0..15
    const uint16_t* bs1 = &Bs[cur][(size_t)(16 + l16) * 32 + hi * 8];  // cols 16..31
    b0.q[0] = *(const uint4*)(bs0);
    b0.q[1] = *(const uint4*)(bs0 + 16);
    b1.q[0] = *(const uint4*)(bs1);
    b1.q[1] = *(const uint4*)(bs1 + 16);

    acc[0][0].v = __builtin_amdgcn_wmma_f32_16x16x32_bf16(
        false, a0.v, false, b0.v, (short)0, acc[0][0].v, false, false);
    acc[0][1].v = __builtin_amdgcn_wmma_f32_16x16x32_bf16(
        false, a0.v, false, b1.v, (short)0, acc[0][1].v, false, false);
    acc[1][0].v = __builtin_amdgcn_wmma_f32_16x16x32_bf16(
        false, a1.v, false, b0.v, (short)0, acc[1][0].v, false, false);
    acc[1][1].v = __builtin_amdgcn_wmma_f32_16x16x32_bf16(
        false, a1.v, false, b1.v, (short)0, acc[1][1].v, false, false);

    __syncthreads();   // waves done reading before next overwrite
    cur ^= 1;
  }

  #pragma unroll
  for (int mi = 0; mi < 2; mi++) {
    const int row0 = mbase + mi * 16 + hi * 8;
    #pragma unroll
    for (int ni = 0; ni < 2; ni++) {
      const int   col = nbase + ni * 16 + l16;
      const float bb  = bias ? bias[col] : 0.f;
      #pragma unroll
      for (int r = 0; r < 8; r++) {
        int    row = row0 + r;
        float  v   = acc[mi][ni].f[r] + bb;
        if (act == 1) v = v / (1.f + __expf(-v));    // silu
        size_t off = (size_t)row * N + col;
        if (mul)  v *= bf2f(mul[off]);
        if (outF) outF[off] = v;
        if (outB) outB[off] = f2bf(v);
      }
    }
  }
}

// ---------------------------------------------------------------------------
// Small utility kernels
// ---------------------------------------------------------------------------
__global__ void f2b_kernel(const float* __restrict__ X, uint16_t* __restrict__ Y, int n) {
  int i = blockIdx.x * blockDim.x + threadIdx.x;
  if (i < n) Y[i] = f2bf(X[i]);
}

__global__ void zero_kernel(float* __restrict__ X, int n) {
  int i = blockIdx.x * blockDim.x + threadIdx.x;
  if (i < n) X[i] = 0.f;
}

__global__ void add_kernel(const float* __restrict__ A, const float* __restrict__ Bv,
                           float* __restrict__ O, uint16_t* __restrict__ Ob, int n) {
  int i = blockIdx.x * blockDim.x + threadIdx.x;
  if (i < n) {
    float v = A[i] + Bv[i];
    O[i] = v;
    if (Ob) Ob[i] = f2bf(v);
  }
}

// silu(fe)[t,j] = silu(log(rho[t]+1e-4) * fe_W[j]) -> bf16 [T,512]
__global__ void silufe_kernel(const float* __restrict__ rho, const float* __restrict__ feW,
                              uint16_t* __restrict__ out, int total) {
  int i = blockIdx.x * blockDim.x + threadIdx.x;
  if (i >= total) return;
  int t = i >> 9, j = i & 511;
  float fe = __logf(rho[t] + 1e-4f) * feW[j];
  out[i] = f2bf(fe / (1.f + __expf(-fe)));
}

// Fourier features: E[t, f]=cos(freq_f . c_t), E[t, F+f]=sin(...), bf16 [T, 2F]
__global__ void fourier_kernel(const float* __restrict__ C, const float* __restrict__ freq,
                               uint16_t* __restrict__ E, int T, int F) {
  int i = blockIdx.x * blockDim.x + threadIdx.x;
  if (i >= T * F) return;
  int t = i / F, f = i % F;
  float cx = C[t * 3 + 0], cy = C[t * 3 + 1], cz = C[t * 3 + 2];
  float fr = freq[f * 3 + 0] * cx + freq[f * 3 + 1] * cy + freq[f * 3 + 2] * cz;
  float s, c;
  __sincosf(fr, &s, &c);
  E[(size_t)t * 2 * F + f]     = f2bf(c);
  E[(size_t)t * 2 * F + F + f] = f2bf(s);
}

// LayerNorm over 128 dims, one wave / token; optional f32 and bf16 outputs.
__global__ __launch_bounds__(256) void ln_kernel(
    const float* __restrict__ X, const float* __restrict__ g, const float* __restrict__ be,
    float* __restrict__ outF, uint16_t* __restrict__ outB, int T)
{
  int t = blockIdx.x * 8 + (threadIdx.x >> 5);
  int lane = threadIdx.x & 31;
  if (t >= T) return;
  const float4 x = *(const float4*)(X + (size_t)t * 128 + lane * 4);
  float mean = wave_sum(x.x + x.y + x.z + x.w) * (1.f / 128.f);
  float d0 = x.x - mean, d1 = x.y - mean, d2 = x.z - mean, d3 = x.w - mean;
  float var = wave_sum(d0 * d0 + d1 * d1 + d2 * d2 + d3 * d3) * (1.f / 128.f);
  float rs = rsqrtf(var + 1e-5f);
  const float4 g4 = *(const float4*)(g + lane * 4);
  const float4 b4 = *(const float4*)(be + lane * 4);
  float y0 = d0 * rs * g4.x + b4.x, y1 = d1 * rs * g4.y + b4.y;
  float y2 = d2 * rs * g4.z + b4.z, y3 = d3 * rs * g4.w + b4.w;
  if (outF) {
    float4 o; o.x = y0; o.y = y1; o.z = y2; o.w = y3;
    *(float4*)(outF + (size_t)t * 128 + lane * 4) = o;
  }
  if (outB) {
    uint2 p;
    p.x = (uint32_t)f2bf(y0) | ((uint32_t)f2bf(y1) << 16);
    p.y = (uint32_t)f2bf(y2) | ((uint32_t)f2bf(y3) << 16);
    *(uint2*)(void*)(outB + (size_t)t * 128 + lane * 4) = p;
  }
}

// ---------------------------------------------------------------------------
// Flash-style attention, dh=16, one wave per (b, h, q); lanes stride keys.
// biasMode: 0 = +log(weights[b,k]); 1 = none; 2 = -softplus(temp[h])*dist*invstd
// Q/K/V/O layout: [b*T + t, 128] with head h at columns [h*16, h*16+16)
// ---------------------------------------------------------------------------
__global__ __launch_bounds__(256) void attn_kernel(
    const float* __restrict__ Q, const float* __restrict__ K, const float* __restrict__ V,
    float* __restrict__ Of, uint16_t* __restrict__ Ob,
    int Bn, int TQ, int TK, int Hn, int biasMode,
    const float* __restrict__ wkeys, const float* __restrict__ dists,
    const float* __restrict__ invstd, const float* __restrict__ temps)
{
  int gw = blockIdx.x * 8 + (threadIdx.x >> 5);
  int lane = threadIdx.x & 31;
  int q = gw % TQ;
  int t = gw / TQ;
  int h = t % Hn;
  int b = t / Hn;

  const float* qp = Q + ((size_t)(b * TQ + q) * 128) + h * 16;
  float qv[16];
  #pragma unroll
  for (int i = 0; i < 16; i++) qv[i] = qp[i];

  float spc = 0.f;
  if (biasMode == 2) spc = softplusf(temps[h]) * invstd[b];

  float mcur = -1e30f, l = 0.f;
  float o[16];
  #pragma unroll
  for (int i = 0; i < 16; i++) o[i] = 0.f;

  for (int k0 = 0; k0 < TK; k0 += 32) {
    int kk = k0 + lane;
    const float* kp = K + ((size_t)(b * TK + kk) * 128) + h * 16;
    float s = 0.f;
    #pragma unroll
    for (int i = 0; i < 16; i++) s += qv[i] * kp[i];
    s *= 0.25f;  // 1/sqrt(16)
    if (biasMode == 0)      s += __logf(wkeys[(size_t)b * TK + kk]);
    else if (biasMode == 2) s -= spc * dists[((size_t)b * TQ + q) * TK + kk];

    float mnew = fmaxf(mcur, wave_max(s));
    float p = __expf(s - mnew);
    float corr = __expf(mcur - mnew);
    mcur = mnew;
    l = l * corr + wave_sum(p);
    const float* vp = V + ((size_t)(b * TK + kk) * 128) + h * 16;
    #pragma unroll
    for (int i = 0; i < 16; i++) o[i] = o[i] * corr + p * vp[i];
  }

  float inv_l = 1.f / l;
  size_t obase = (size_t)(b * TQ + q) * 128 + h * 16;
  #pragma unroll
  for (int i = 0; i < 16; i++) {
    float s = wave_sum(o[i]);
    if (lane == 0) {
      float r = s * inv_l;
      Of[obase + i] = r;
      if (Ob) Ob[obase + i] = f2bf(r);
    }
  }
}

// ---------------------------------------------------------------------------
// Per-batch weighted mean/cov + analytic 3x3 eigenvalues -> scale[b][3]
// ---------------------------------------------------------------------------
__device__ inline void eig3_sym(float a00, float a01, float a02, float a11, float a12, float a22,
                                float& e0, float& e1, float& e2)
{
  float p1 = a01 * a01 + a02 * a02 + a12 * a12;
  if (p1 < 1e-30f) { e0 = a00; e1 = a11; e2 = a22; }
  else {
    float q = (a00 + a11 + a22) * (1.f / 3.f);
    float p2 = (a00 - q) * (a00 - q) + (a11 - q) * (a11 - q) + (a22 - q) * (a22 - q) + 2.f * p1;
    float p = sqrtf(p2 * (1.f / 6.f));
    float ip = 1.f / p;
    float b00 = (a00 - q) * ip, b11 = (a11 - q) * ip, b22 = (a22 - q) * ip;
    float b01 = a01 * ip, b02 = a02 * ip, b12 = a12 * ip;
    float detB = b00 * (b11 * b22 - b12 * b12) - b01 * (b01 * b22 - b12 * b02)
               + b02 * (b01 * b12 - b11 * b02);
    float r = fminf(1.f, fmaxf(-1.f, 0.5f * detB));
    float phi = acosf(r) * (1.f / 3.f);
    float big = q + 2.f * p * cosf(phi);
    float sml = q + 2.f * p * cosf(phi + 2.09439510239f);
    e0 = sml; e1 = 3.f * q - big - sml; e2 = big;
  }
  float tv;
  if (e0 > e1) { tv = e0; e0 = e1; e1 = tv; }
  if (e1 > e2) { tv = e1; e1 = e2; e2 = tv; }
  if (e0 > e1) { tv = e0; e0 = e1; e1 = tv; }
}

__global__ __launch_bounds__(256) void stats_kernel(
    const float* __restrict__ rho, const float* __restrict__ coords,
    const float* __restrict__ w, float* __restrict__ scaleOut, int Nn)
{
  __shared__ float red[10 * 256];
  int b = blockIdx.x, tid = threadIdx.x;
  float s[10];
  #pragma unroll
  for (int i = 0; i < 10; i++) s[i] = 0.f;
  for (int n = tid; n < Nn; n += 256) {
    float wr = w[b * Nn + n] * rho[b * Nn + n];
    const float* c = coords + (size_t)(b * Nn + n) * 3;
    float x = c[0], y = c[1], z = c[2];
    s[0] += wr;
    s[1] += wr * x; s[2] += wr * y; s[3] += wr * z;
    s[4] += wr * x * x; s[5] += wr * x * y; s[6] += wr * x * z;
    s[7] += wr * y * y; s[8] += wr * y * z; s[9] += wr * z * z;
  }
  #pragma unroll
  for (int i = 0; i < 10; i++) red[i * 256 + tid] = s[i];
  __syncthreads();
  for (int off = 128; off > 0; off >>= 1) {
    if (tid < off)
      #pragma unroll
      for (int i = 0; i < 10; i++) red[i * 256 + tid] += red[i * 256 + tid + off];
    __syncthreads();
  }
  if (tid == 0) {
    float S0 = red[0], inv = 1.f / S0;
    float mx = red[1 * 256] * inv, my = red[2 * 256] * inv, mz = red[3 * 256] * inv;
    float cxx = red[4 * 256] * inv - mx * mx;
    float cxy = red[5 * 256] * inv - mx * my;
    float cxz = red[6 * 256] * inv - mx * mz;
    float cyy = red[7 * 256] * inv - my * my;
    float cyz = red[8 * 256] * inv - my * mz;
    float czz = red[9 * 256] * inv - mz * mz;
    float e0, e1, e2;
    eig3_sym(cxx, cxy, cxz, cyy, cyz, czz, e0, e1, e2);
    scaleOut[b * 3 + 0] = 2.f * sqrtf(e0 + 1e-5f);
    scaleOut[b * 3 + 1] = 2.f * sqrtf(e1 + 1e-5f);
    scaleOut[b * 3 + 2] = 2.f * sqrtf(e2 + 1e-5f);
  }
}

__global__ void anchors_kernel(const float* __restrict__ scale, const float* __restrict__ grid,
                               float* __restrict__ anchors, int Bn, int Gn) {
  int i = blockIdx.x * blockDim.x + threadIdx.x;
  if (i >= Bn * Gn * 3) return;
  int b = i / (Gn * 3), r = i % (Gn * 3);
  anchors[i] = scale[b * 3 + (r % 3)] * grid[r];
}

// dists[b,qi,kj] = |anchors[b,qi] - (anchors[b,kj]+1e-5)| ; accumulate sum/sumsq
__global__ __launch_bounds__(256) void dist_kernel(
    const float* __restrict__ A, float* __restrict__ Dst, float* __restrict__ sums, int Gn)
{
  __shared__ float rs[256], rq[256];
  int i = blockIdx.x * 256 + threadIdx.x;
  int per = Gn * Gn;
  int b = i / per, r = i % per;
  int qi = r / Gn, kj = r % Gn;
  const float* aq = A + (size_t)(b * Gn + qi) * 3;
  const float* ak = A + (size_t)(b * Gn + kj) * 3;
  float dx = aq[0] - (ak[0] + 1e-5f);
  float dy = aq[1] - (ak[1] + 1e-5f);
  float dz = aq[2] - (ak[2] + 1e-5f);
  float d = sqrtf(dx * dx + dy * dy + dz * dz);
  Dst[i] = d;
  int tid = threadIdx.x;
  rs[tid] = d; rq[tid] = d * d;
  __syncthreads();
  for (int off = 128; off > 0; off >>= 1) {
    if (tid < off) { rs[tid] += rs[tid + off]; rq[tid] += rq[tid + off]; }
    __syncthreads();
  }
  if (tid == 0) {
    atomicAdd(&sums[b * 2 + 0], rs[0]);
    atomicAdd(&sums[b * 2 + 1], rq[0]);
  }
}

__global__ void invstd_kernel(const float* __restrict__ sums, float* __restrict__ invstd, int Gn) {
  int b = threadIdx.x;
  if (b >= 4) return;
  float n = (float)Gn * (float)Gn;
  float mean = sums[b * 2] / n;
  float var = sums[b * 2 + 1] / n - mean * mean;
  invstd[b] = rsqrtf(fmaxf(var, 1e-30f));
}

// head: out[t] = softplus(x.W0+b0) * lda(clip(rho)) - log_cosh(x.W1+b1)
__global__ __launch_bounds__(256) void head_final_kernel(
    const float* __restrict__ X, const float* __restrict__ W, const float* __restrict__ bb,
    const float* __restrict__ rho, float* __restrict__ out, int T)
{
  int t = blockIdx.x * 8 + (threadIdx.x >> 5);
  int lane = threadIdx.x & 31;
  const float4 x = *(const float4*)(X + (size_t)t * 128 + lane * 4);
  const float4 w0 = *(const float4*)(W + lane * 4);
  const float4 w1 = *(const float4*)(W + 128 + lane * 4);
  float d0 = wave_sum(x.x * w0.x + x.y * w0.y + x.z * w0.z + x.w * w0.w);
  float d1 = wave_sum(x.x * w1.x + x.y * w1.y + x.z * w1.z + x.w * w1.w);
  if (lane == 0) {
    float o0 = d0 + bb[0], o1 = d1 + bb[1];
    float scale = softplusf(o0);
    float lc = o1 + softplusf(-2.f * o1) - 0.69314718056f;   // log_cosh
    float r = fmaxf(rho[t], 1e-7f);
    out[t] = scale * (-0.73855876668f * cbrtf(r)) - lc;
  }
}

// ---------------------------------------------------------------------------
// Host orchestration
// ---------------------------------------------------------------------------
extern "C" void kernel_launch(void* const* d_in, const int* in_sizes, int n_in,
                              void* d_out, int out_size, void* d_ws, size_t ws_size,
                              hipStream_t stream)
{
  (void)in_sizes; (void)n_in; (void)out_size; (void)ws_size;
  const int Bb = 4, Nn = 4096, Gg = 512, Hh = 8;
  const int Tp = Bb * Nn;   // 16384 point tokens
  const int Tg = Bb * Gg;   // 2048 grid tokens

  auto F = [&](int i) { return (const float*)d_in[i]; };

  // ---- input leaves (top-level insertion order, params = sorted pytree) ----
  const float* rho    = F(0);
  const float* coords = F(1);
  const float* wts    = F(2);
  const float* gridc  = F(3);
  struct BlkP { const float *Wk,*Wo,*Wq,*Wv,*bk,*bo,*bq,*bv,*temp,*ln1b,*ln1g,*ln2b,*ln2g,*mW1,*mW2,*mb1,*mb2; };
  BlkP blk[2];
  for (int i = 0; i < 2; i++) {
    int o = 4 + i * 17;
    blk[i] = { F(o+0),F(o+1),F(o+2),F(o+3),F(o+4),F(o+5),F(o+6),F(o+7),F(o+8),
               F(o+9),F(o+10),F(o+11),F(o+12),F(o+13),F(o+14),F(o+15),F(o+16) };
  }
  const float* feW    = F(38);
  const float* ceFreq = F(39); const float* ceW1 = F(40); const float* ceW2 = F(41);
  const float* ceb1   = F(42); const float* ceb2 = F(43);
  const float* projW  = F(44); const float* projb = F(45);
  const float* geFreq = F(46); const float* geW1 = F(47); const float* geW2 = F(48);
  const float* geb1   = F(49); const float* geb2 = F(50);
  const float* headW  = F(51); const float* headb = F(52);
  const float* hlnb   = F(53); const float* hlng = F(54);
  const float* hW1    = F(55); const float* hW2 = F(56);
  const float* hb1    = F(57); const float* hb2 = F(58);
  struct InterpP { const float *Wk,*Wo,*Wq,*Wv,*bk,*bo,*bq,*bv,*lnkb,*lnkg,*lnqb,*lnqg; };
  InterpP ii = { F(59),F(60),F(61),F(62),F(63),F(64),F(65),F(66),F(67),F(68),F(69),F(70) };
  InterpP io = { F(71),F(72),F(73),F(74),F(75),F(76),F(77),F(78),F(79),F(80),F(81),F(82) };

  // ---- workspace allocator ----
  char* wsb = (char*)d_ws; size_t cur = 0;
  auto alloc = [&](size_t bytes) -> void* {
    cur = (cur + 255) & ~(size_t)255;
    void* p = wsb + cur; cur += bytes; return p;
  };
  auto af = [&](size_t n) { return (float*)alloc(n * 4); };
  auto ab = [&](size_t n) { return (uint16_t*)alloc(n * 2); };

  // bf16 weight copies
  uint16_t* ceW1b = ab(512 * 512); uint16_t* ceW2b = ab(512 * 512);
  uint16_t* projWb = ab(128 * 512);
  uint16_t* geW1b = ab(128 * 128); uint16_t* geW2b = ab(128 * 128);
  uint16_t* iiWb[4]; for (int i = 0; i < 4; i++) iiWb[i] = ab(128 * 128); // q,k,v,o
  uint16_t* ioWb[4]; for (int i = 0; i < 4; i++) ioWb[i] = ab(128 * 128);
  uint16_t* bWb[2][4]; uint16_t* bM1b[2]; uint16_t* bM2b[2];
  for (int i = 0; i < 2; i++) {
    for (int j = 0; j < 4; j++) bWb[i][j] = ab(128 * 128);
    bM1b[i] = ab(512 * 128); bM2b[i] = ab(128 * 512);
  }
  uint16_t* hW1b = ab(512 * 128); uint16_t* hW2b = ab(128 * 512);

  // activations
  uint16_t* Ebuf  = ab((size_t)Tp * 512);   // fourier E, later reused for X
  uint16_t* H1buf = ab((size_t)Tp * 512);   // ce hidden / head hidden
  uint16_t* SFbuf = ab((size_t)Tp * 512);   // silu(fe)
  float*    PHI   = af((size_t)Tp * 128);
  uint16_t* PHIb  = ab((size_t)Tp * 128);
  uint16_t* GEe   = ab((size_t)Tg * 128);
  uint16_t* GEh1  = ab((size_t)Tg * 128);
  float*    GEMB  = af((size_t)Tg * 128);
  uint16_t* LNA   = ab((size_t)Tp * 128);
  uint16_t* LNB   = ab((size_t)Tg * 128);
  float*    Qb    = af((size_t)Tp * 128);
  float*    Kb    = af((size_t)Tp * 128);
  float*    Vb    = af((size_t)Tp * 128);
  float*    Ofb   = af((size_t)Tp * 128);
  uint16_t* Obb   = ab((size_t)Tp * 128);
  float*    Pb    = af((size_t)Tp * 128);
  float*    PG    = af((size_t)Tg * 128);
  uint16_t* PGb   = ab((size_t)Tg * 128);
  uint16_t* M1b   = ab((size_t)Tg * 512);
  float*    LNF   = af((size_t)Tg * 128);
  float*    PHIPTS= af((size_t)Tp * 128);
  float*    dists = af((size_t)Bb * Gg * Gg);
  float*    anchors = af((size_t)Tg * 3);
  float*    scaleB  = af(Bb * 3);
  float*    sums    = af(Bb * 2);
  float*    invstd  = af(Bb);

  // ---- helpers ----
  auto conv = [&](const float* src, uint16_t* dst, int n) {
    f2b_kernel<<<(n + 255) / 256, 256, 0, stream>>>(src, dst, n);
  };
  auto gemm = [&](const uint16_t* A, const uint16_t* W, const float* bias, const uint16_t* mul,
                  float* oF, uint16_t* oB, int M, int N, int K, int act) {
    dim3 g(N / 32, M / 128);
    gemm_bf16_wmma<<<g, 128, 0, stream>>>(A, W, bias, mul, oF, oB, M, N, K, act);
  };
  auto ln = [&](const float* X, const float* g, const float* b, float* oF, uint16_t* oB, int T) {
    ln_kernel<<<T / 8, 256, 0, stream>>>(X, g, b, oF, oB, T);
  };
  auto attn = [&](const float* Q, const float* K, const float* V, float* Of, uint16_t* Ob,
                  int TQ, int TK, int mode, const float* wk, const float* ds,
                  const float* is, const float* tm) {
    int waves = Bb * Hh * TQ;
    attn_kernel<<<waves / 8, 256, 0, stream>>>(Q, K, V, Of, Ob, Bb, TQ, TK, Hh, mode, wk, ds, is, tm);
  };

  // ---- weight conversion ----
  conv(ceW1, ceW1b, 512 * 512); conv(ceW2, ceW2b, 512 * 512);
  conv(projW, projWb, 128 * 512);
  conv(geW1, geW1b, 128 * 128); conv(geW2, geW2b, 128 * 128);
  conv(ii.Wq, iiWb[0], 128 * 128); conv(ii.Wk, iiWb[1], 128 * 128);
  conv(ii.Wv, iiWb[2], 128 * 128); conv(ii.Wo, iiWb[3], 128 * 128);
  conv(io.Wq, ioWb[0], 128 * 128); conv(io.Wk, ioWb[1], 128 * 128);
  conv(io.Wv, ioWb[2], 128 * 128); conv(io.Wo, ioWb[3], 128 * 128);
  for (int i = 0; i < 2; i++) {
    conv(blk[i].Wq, bWb[i][0], 128 * 128); conv(blk[i].Wk, bWb[i][1], 128 * 128);
    conv(blk[i].Wv, bWb[i][2], 128 * 128); conv(blk[i].Wo, bWb[i][3], 128 * 128);
    conv(blk[i].mW1, bM1b[i], 512 * 128);  conv(blk[i].mW2, bM2b[i], 128 * 512);
  }
  conv(hW1, hW1b, 512 * 128); conv(hW2, hW2b, 128 * 512);

  // ---- feature embedding of points ----
  fourier_kernel<<<(Tp * 256 + 255) / 256, 256, 0, stream>>>(coords, ceFreq, Ebuf, Tp, 256);
  silufe_kernel<<<(Tp * 512 + 255) / 256, 256, 0, stream>>>(rho, feW, SFbuf, Tp * 512);
  gemm(Ebuf, ceW1b, ceb1, nullptr, nullptr, H1buf, Tp, 512, 512, 1);         // h1 = silu(E W1 + b1)
  gemm(H1buf, ceW2b, ceb2, SFbuf, nullptr, Ebuf, Tp, 512, 512, 0);           // X = silu(fe) * ce
  gemm(Ebuf, projWb, projb, nullptr, PHI, PHIb, Tp, 128, 512, 0);            // phi

  // ---- grid setup ----
  stats_kernel<<<Bb, 256, 0, stream>>>(rho, coords, wts, scaleB, Nn);
  anchors_kernel<<<(Tg * 3 + 255) / 256, 256, 0, stream>>>(scaleB, gridc, anchors, Bb, Gg);
  fourier_kernel<<<(Tg * 64 + 255) / 256, 256, 0, stream>>>(anchors, geFreq, GEe, Tg, 64);
  gemm(GEe, geW1b, geb1, nullptr, nullptr, GEh1, Tg, 128, 128, 1);
  gemm(GEh1, geW2b, geb2, nullptr, GEMB, nullptr, Tg, 128, 128, 0);          // grid_emb

  // ---- interp_in: grid queries attend to points ----
  ln(GEMB, ii.lnqg, ii.lnqb, nullptr, LNB, Tg);
  ln(PHI,  ii.lnkg, ii.lnkb, nullptr, LNA, Tp);
  gemm(LNB,  iiWb[0], ii.bq, nullptr, Qb, nullptr, Tg, 128, 128, 0);
  gemm(LNA,  iiWb[1], ii.bk, nullptr, Kb, nullptr, Tp, 128, 128, 0);
  gemm(PHIb, iiWb[2], ii.bv, nullptr, Vb, nullptr, Tp, 128, 128, 0);
  attn(Qb, Kb, Vb, Ofb, Obb, Gg, Nn, 0, wts, nullptr, nullptr, nullptr);
  gemm(Obb, iiWb[3], ii.bo, nullptr, PG, PGb, Tg, 128, 128, 0);              // phi_grid

  // ---- dists + std normalization ----
  zero_kernel<<<1, 256, 0, stream>>>(sums, Bb * 2);
  dist_kernel<<<(Bb * Gg * Gg) / 256, 256, 0, stream>>>(anchors, dists, sums, Gg);
  invstd_kernel<<<1, 64, 0, stream>>>(sums, invstd, Gg);

  // ---- interaction blocks ----
  for (int i = 0; i < 2; i++) {
    ln(PG, blk[i].ln1g, blk[i].ln1b, nullptr, LNB, Tg);
    gemm(LNB, bWb[i][0], blk[i].bq, nullptr, Qb, nullptr, Tg, 128, 128, 0);
    gemm(LNB, bWb[i][1], blk[i].bk, nullptr, Kb, nullptr, Tg, 128, 128, 0);
    gemm(LNB, bWb[i][2], blk[i].bv, nullptr, Vb, nullptr, Tg, 128, 128, 0);
    attn(Qb, Kb, Vb, Ofb, Obb, Gg, Gg, 2, nullptr, dists, invstd, blk[i].temp);
    gemm(Obb, bWb[i][3], blk[i].bo, nullptr, Pb, nullptr, Tg, 128, 128, 0);
    add_kernel<<<(Tg * 128 + 255) / 256, 256, 0, stream>>>(PG, Pb, PG, PGb, Tg * 128);
    gemm(PGb, bM1b[i], blk[i].mb1, nullptr, nullptr, M1b, Tg, 512, 128, 1);
    gemm(M1b, bM2b[i], blk[i].mb2, nullptr, Pb, nullptr, Tg, 128, 512, 0);
    ln(Pb, blk[i].ln2g, blk[i].ln2b, LNF, nullptr, Tg);
    add_kernel<<<(Tg * 128 + 255) / 256, 256, 0, stream>>>(PG, LNF, PG, PGb, Tg * 128);
  }

  // ---- interp_out: point queries attend to grid ----
  ln(PHI, io.lnqg, io.lnqb, nullptr, LNA, Tp);
  ln(PG,  io.lnkg, io.lnkb, nullptr, LNB, Tg);
  gemm(LNA, ioWb[0], io.bq, nullptr, Qb, nullptr, Tp, 128, 128, 0);
  gemm(LNB, ioWb[1], io.bk, nullptr, Kb, nullptr, Tg, 128, 128, 0);
  gemm(PGb, ioWb[2], io.bv, nullptr, Vb, nullptr, Tg, 128, 128, 0);
  attn(Qb, Kb, Vb, Ofb, Obb, Nn, Gg, 1, nullptr, nullptr, nullptr, nullptr);
  gemm(Obb, ioWb[3], io.bo, nullptr, Pb, nullptr, Tp, 128, 128, 0);
  add_kernel<<<(Tp * 128 + 255) / 256, 256, 0, stream>>>(PHI, Pb, PHIPTS, nullptr, Tp * 128);

  // ---- head ----
  ln(PHIPTS, hlng, hlnb, nullptr, LNA, Tp);
  gemm(LNA, hW1b, hb1, nullptr, nullptr, H1buf, Tp, 512, 128, 1);
  gemm(H1buf, hW2b, hb2, nullptr, Pb, nullptr, Tp, 128, 512, 0);
  head_final_kernel<<<Tp / 8, 256, 0, stream>>>(Pb, headW, headb, rho, (float*)d_out, Tp);
}
